// GeodesicAttention_13623636262999
// MI455X (gfx1250) — matrix operand order repaired and checked by reading, not compiled, not run on hardware
//
#include <hip/hip_runtime.h>

// GeodesicAttention for MI455X (gfx1250, wave32, WMMA f32_16x16x32_f16).
//
// softmax(-dist^2) == softmax(2*y_t.y_j - n_j)  (per-row shift invariance),
// so with ys = sqrt(2)*y (f16) the score tile is one WMMA with C = -n_j.
// Dominant cost: out = P @ x = 34.4 GFLOP -> f16 WMMA, fp32 accum,
// 2x2 register-blocked (4 independent accumulators per wave).
// All intermediates (~84.5 MB ws) + io fit in 192 MB L2.

typedef __attribute__((ext_vector_type(16))) _Float16 v16h;
typedef __attribute__((ext_vector_type(8)))  _Float16 v8h;
typedef __attribute__((ext_vector_type(8)))  float    v8f;
typedef __attribute__((ext_vector_type(4)))  float    v4f;

#define B_ 4
#define T_ 2048
#define D_ 1024
#define R_ 32

// ---------------- kernel 0: x (fp32 [B,T,D]) -> xhT (f16 [B,D,T]) ----------
__global__ void ga_transpose(const float* __restrict__ x, _Float16* __restrict__ xhT) {
  __shared__ _Float16 tile[32][33];
  const int t0 = blockIdx.x * 32, d0 = blockIdx.y * 32, b = blockIdx.z;
  const float* xb = x + (size_t)b * T_ * D_;
  _Float16* xo = xhT + (size_t)b * D_ * T_;
  const int tx = threadIdx.x, ty = threadIdx.y;
#pragma unroll
  for (int i = 0; i < 4; ++i) {
    int t = t0 + ty + i * 8;
    tile[ty + i * 8][tx] = (_Float16)xb[(size_t)t * D_ + d0 + tx];
  }
  __syncthreads();
#pragma unroll
  for (int i = 0; i < 4; ++i) {
    int d = d0 + ty + i * 8;
    xo[(size_t)d * T_ + t0 + tx] = tile[tx][ty + i * 8];
  }
}

// ------------- kernel 0b: prepack U into per-lane WMMA B fragments ---------
// uf[nt][kc][lane] : v16h, halves h -> B[k = (lane>>4)*16 + h][n = nt*16 + (lane&15)]
__global__ void ga_upack(const float* __restrict__ U, _Float16* __restrict__ uf) {
  int id = blockIdx.x * 256 + threadIdx.x;   // 0..2047
  if (id >= 2048) return;
  int l  = id & 31;
  int kc = (id >> 5) & 31;
  int nt = id >> 10;
  int n  = nt * 16 + (l & 15);
  int kg = kc * 32 + (l >> 4) * 16;
  v16h v;
#pragma unroll
  for (int h = 0; h < 16; ++h) v[h] = (_Float16)U[(size_t)(kg + h) * R_ + n];
  *(v16h*)(uf + (size_t)id * 16) = v;
}

// --------- kernel 1: y = x @ U via WMMA; emit ys = sqrt2*y (f16), n --------
__global__ void ga_project(const float* __restrict__ x, const _Float16* __restrict__ uf,
                           _Float16* __restrict__ ys, float* __restrict__ nn) {
  const int lane = threadIdx.x & 31;
  const int w    = threadIdx.x >> 5;
  const int g    = blockIdx.x * 4 + w;      // tile 0..511, bt0 = g*16
  const int bt0  = g * 16;
  const int hi   = lane >> 4, lp = lane & 15;
  const float* arow = x + (size_t)(bt0 + lp) * D_;
  v8f acc0 = {}, acc1 = {};
  for (int kc = 0; kc < 32; ++kc) {
    int koff = kc * 32 + hi * 16;
    v4f f0 = *(const v4f*)(arow + koff);
    v4f f1 = *(const v4f*)(arow + koff + 4);
    v4f f2 = *(const v4f*)(arow + koff + 8);
    v4f f3 = *(const v4f*)(arow + koff + 12);
    v16h a;
#pragma unroll
    for (int i = 0; i < 4; ++i) {
      a[i] = (_Float16)f0[i]; a[4 + i] = (_Float16)f1[i];
      a[8 + i] = (_Float16)f2[i]; a[12 + i] = (_Float16)f3[i];
    }
    v16h b0 = *(const v16h*)(uf + ((size_t)(kc)*32 + lane) * 16);
    v16h b1 = *(const v16h*)(uf + ((size_t)(32 + kc) * 32 + lane) * 16);
    acc0 = __builtin_amdgcn_wmma_f32_16x16x32_f16(false, a, false, b0, (short)0, acc0, false, false);
    acc1 = __builtin_amdgcn_wmma_f32_16x16x32_f16(false, a, false, b1, (short)0, acc1, false, false);
  }
  const float s2 = 1.41421356237309515f;
#pragma unroll
  for (int v = 0; v < 8; ++v) {
    int row = bt0 + v + hi * 8;              // C/D layout: M = v + 8*hi, N = lp
    ys[(size_t)row * 32 + lp]      = (_Float16)(acc0[v] * s2);
    ys[(size_t)row * 32 + 16 + lp] = (_Float16)(acc1[v] * s2);
  }
#pragma unroll
  for (int v = 0; v < 8; ++v) {              // n[row] = sum_r y^2 : reduce 16 lanes
    float p = acc0[v] * acc0[v] + acc1[v] * acc1[v];
    p += __shfl_xor(p, 1, 32);
    p += __shfl_xor(p, 2, 32);
    p += __shfl_xor(p, 4, 32);
    p += __shfl_xor(p, 8, 32);
    if (lp == 0) nn[bt0 + v + hi * 8] = p;
  }
}

// ----------- kernel 2: S[bt, j] = ys_t . ys_j - n_j  (one WMMA/tile) -------
__global__ void ga_scores(const _Float16* __restrict__ ys, const float* __restrict__ nn,
                          float* __restrict__ S) {
  const int lane = threadIdx.x & 31;
  const int w    = threadIdx.x >> 5;
  const int g    = blockIdx.x * 4 + w;      // 0..511
  const int b    = g >> 7;
  const int bt0  = g * 16;
  const int hi   = lane >> 4, lp = lane & 15;
  // A fragment: rows of ys for this t-tile (K = R = 32, single chunk)
  const _Float16* ar = ys + (size_t)(bt0 + lp) * 32;
  v8h alo = *(const v8h*)(ar + hi * 16);
  v8h ahi = *(const v8h*)(ar + hi * 16 + 8);
  v16h a = __builtin_shufflevector(alo, ahi, 0, 1, 2, 3, 4, 5, 6, 7,
                                   8, 9, 10, 11, 12, 13, 14, 15);
  const int jrow0 = b * T_;
  for (int jt = 0; jt < 128; ++jt) {
    int j0 = jt * 16;
    int jrow = jrow0 + j0;
    v16h bf = *(const v16h*)(ys + (size_t)(jrow + lp) * 32 + hi * 16);
    float nv = -nn[jrow + lp];               // C[m][n] = -n_j, constant down column
    v8f c = {nv, nv, nv, nv, nv, nv, nv, nv};
    v8f d = __builtin_amdgcn_wmma_f32_16x16x32_f16(false, a, false, bf, (short)0, c, false, false);
#pragma unroll
    for (int v = 0; v < 8; ++v)
      S[(size_t)(bt0 + v + hi * 8) * T_ + j0 + lp] = d[v];
  }
}

// ------ kernel 3: row softmax; write P as f16 in place over the row --------
__global__ void ga_softmax(float* __restrict__ S) {
  __shared__ float red[256];
  const size_t row = blockIdx.x;            // 0..B*T-1
  float* srow = S + row * T_;
  const int tid = threadIdx.x;
  float vals[8];
#pragma unroll
  for (int k = 0; k < 8; ++k) vals[k] = srow[tid + k * 256];
  float m = -1e30f;
#pragma unroll
  for (int k = 0; k < 8; ++k) m = fmaxf(m, vals[k]);
  red[tid] = m;
  __syncthreads();                           // all reads of srow complete here
  for (int s = 128; s > 0; s >>= 1) {
    if (tid < s) red[tid] = fmaxf(red[tid], red[tid + s]);
    __syncthreads();
  }
  m = red[0];
  __syncthreads();
  float e[8], sum = 0.f;
#pragma unroll
  for (int k = 0; k < 8; ++k) { e[k] = __expf(vals[k] - m); sum += e[k]; }
  red[tid] = sum;
  __syncthreads();
  for (int s = 128; s > 0; s >>= 1) {
    if (tid < s) red[tid] += red[tid + s];
    __syncthreads();
  }
  float inv = 1.0f / red[0];
  _Float16* prow = (_Float16*)srow;          // f16 P occupies first 2*T bytes of row
#pragma unroll
  for (int k = 0; k < 8; ++k) prow[tid + k * 256] = (_Float16)(e[k] * inv);
}

// --------- kernel 4: out = P @ x  (34.4 GFLOP, f16 WMMA, K = T = 2048) -----
// 2x2 register blocking: each wave owns a 32(t) x 32(n) output, i.e. four
// 16x16 accumulators; A fragments reused across n-tiles, B across t-tiles.
// Block = 8 waves in 2x4 -> 64(t) x 128(n) tile. Grid = B*(T/64)*(D/128).
__global__ void ga_out(const float* __restrict__ S, const _Float16* __restrict__ xhT,
                       float* __restrict__ out) {
  const int lane = threadIdx.x & 31;
  const int w    = threadIdx.x >> 5;        // 0..7
  const int bi   = blockIdx.x;
  const int b    = bi >> 8;                 // 256 blocks per batch
  const int rem  = bi & 255;
  const int tb   = rem >> 3;                // 0..31 (t blocks of 64)
  const int nb   = rem & 7;                 // 0..7  (n blocks of 128)
  const int t0   = tb * 64 + (w & 1) * 32;
  const int n0   = nb * 128 + (w >> 1) * 32;
  const int hi   = lane >> 4, lp = lane & 15;

  const _Float16* prow0 = (const _Float16*)(S + (size_t)(b * T_ + t0 + lp) * T_);
  const _Float16* prow1 = (const _Float16*)(S + (size_t)(b * T_ + t0 + 16 + lp) * T_);
  const _Float16* bcol0 = xhT + ((size_t)b * D_ + n0 + lp) * T_ + hi * 16;
  const _Float16* bcol1 = xhT + ((size_t)b * D_ + n0 + 16 + lp) * T_ + hi * 16;

  v8f acc00 = {}, acc01 = {}, acc10 = {}, acc11 = {};
  for (int kc = 0; kc < 64; ++kc) {
    int koff = kc * 32 + hi * 16;
    v8h a0lo = *(const v8h*)(prow0 + koff);
    v8h a0hi = *(const v8h*)(prow0 + koff + 8);
    v8h a1lo = *(const v8h*)(prow1 + koff);
    v8h a1hi = *(const v8h*)(prow1 + koff + 8);
    v16h a0 = __builtin_shufflevector(a0lo, a0hi, 0, 1, 2, 3, 4, 5, 6, 7,
                                      8, 9, 10, 11, 12, 13, 14, 15);
    v16h a1 = __builtin_shufflevector(a1lo, a1hi, 0, 1, 2, 3, 4, 5, 6, 7,
                                      8, 9, 10, 11, 12, 13, 14, 15);
    v16h b0 = *(const v16h*)(bcol0 + kc * 32);   // xhT: K contiguous per lane
    v16h b1 = *(const v16h*)(bcol1 + kc * 32);
    acc00 = __builtin_amdgcn_wmma_f32_16x16x32_f16(false, a0, false, b0, (short)0, acc00, false, false);
    acc01 = __builtin_amdgcn_wmma_f32_16x16x32_f16(false, a0, false, b1, (short)0, acc01, false, false);
    acc10 = __builtin_amdgcn_wmma_f32_16x16x32_f16(false, a1, false, b0, (short)0, acc10, false, false);
    acc11 = __builtin_amdgcn_wmma_f32_16x16x32_f16(false, a1, false, b1, (short)0, acc11, false, false);
  }
#pragma unroll
  for (int v = 0; v < 8; ++v) {
    size_t r0 = (size_t)(b * T_ + t0 + v + hi * 8) * D_;
    size_t r1 = (size_t)(b * T_ + t0 + 16 + v + hi * 8) * D_;
    out[r0 + n0 + lp]      = acc00[v];
    out[r0 + n0 + 16 + lp] = acc01[v];
    out[r1 + n0 + lp]      = acc10[v];
    out[r1 + n0 + 16 + lp] = acc11[v];
  }
}

// ---------------------------------------------------------------------------
extern "C" void kernel_launch(void* const* d_in, const int* in_sizes, int n_in,
                              void* d_out, int out_size, void* d_ws, size_t ws_size,
                              hipStream_t stream) {
  const float* x = (const float*)d_in[0];   // [B,T,D] fp32
  const float* U = (const float*)d_in[1];   // [D,R]   fp32
  float* out = (float*)d_out;               // [B,T,D] fp32

  char* ws = (char*)d_ws;
  // workspace layout (all 256B aligned)
  _Float16* xhT = (_Float16*)(ws);                                   // 16,777,216 B
  _Float16* ys  = (_Float16*)(ws + 16777216);                        //    524,288 B
  float*    nn  = (float*)   (ws + 16777216 + 524288);               //     32,768 B
  _Float16* uf  = (_Float16*)(ws + 16777216 + 524288 + 32768);       //     65,536 B
  float*    Smat= (float*)   (ws + 16777216 + 524288 + 32768 + 65536); // 67,108,864 B
  // total: 84,508,672 bytes

  ga_transpose<<<dim3(T_ / 32, D_ / 32, B_), dim3(32, 8), 0, stream>>>(x, xhT);
  ga_upack<<<8, 256, 0, stream>>>(U, uf);
  ga_project<<<(B_ * T_) / 64, 128, 0, stream>>>(x, uf, ys, nn);
  ga_scores<<<(B_ * T_) / 64, 128, 0, stream>>>(ys, nn, Smat);
  ga_softmax<<<B_ * T_, 256, 0, stream>>>(Smat);
  ga_out<<<B_ * (T_ / 64) * (D_ / 128), 256, 0, stream>>>(Smat, xhT, out);
}
